// GemmaAttention_89936615178242
// MI455X (gfx1250) — matrix operand order repaired and checked by reading, not compiled, not run on hardware
//
#include <hip/hip_runtime.h>

// ---------------------------------------------------------------------------
// Problem constants
// ---------------------------------------------------------------------------
#define HID   3584
#define NH    16
#define NKV   8
#define HD    256
#define SEQ   3072
#define WIN   2048
#define NQKV  8192          // NH*HD + 2*NKV*HD
#define QOFF  0
#define KOFF  4096
#define VOFF  6144
#define SOFTCAP 50.0f
#define QSCALE  0.0625f     // 256^-0.5
#define NEGF   -2.3819763e+38f

typedef __attribute__((ext_vector_type(16))) __bf16 v16bf;
typedef __attribute__((ext_vector_type(8)))  float  v8f;

union Frag16 {
    v16bf v;
    uint4 u[2];
    unsigned short h[16];
};

__device__ inline unsigned short f2bf(float f) {            // RNE float -> bf16 bits
    unsigned x = __float_as_uint(f);
    x += 0x7fffu + ((x >> 16) & 1u);
    return (unsigned short)(x >> 16);
}
__device__ inline float bf2f(unsigned short h) {
    return __uint_as_float(((unsigned)h) << 16);
}
__device__ inline float tanh_fast(float x) {
    x = fminf(fmaxf(x, -15.f), 15.f);
    float e = __expf(2.f * x);
    return (e - 1.f) / (e + 1.f);
}
__device__ inline v8f wmma_bf16(const Frag16& a, const Frag16& b, v8f c) {
    return __builtin_amdgcn_wmma_f32_16x16x32_bf16(false, a.v, false, b.v,
                                                   (short)0, c, false, false);
}

// ---------------------------------------------------------------------------
// Tiled GEMM: C(MxN) = A(MxK) * B(NxK)^T, bf16 WMMA, fp32 accumulate.
// B is always fp32 in global (weights), converted to bf16 while staging LDS.
// A is fp32 (first GEMM) or bf16 (second GEMM).  Block tile 128x128, kstep 32.
// 256 threads = 8 waves; wave tile 64Mx32N (4x2 frags).
// ---------------------------------------------------------------------------
template <bool A_BF16, bool OUT_BF16>
__global__ __launch_bounds__(256) void gemm_tn(const void* __restrict__ Ap,
                                               const float* __restrict__ Bw,
                                               void* __restrict__ Cp,
                                               int M, int N, int K) {
    __shared__ unsigned short sA[128 * 40];   // rows padded: 32 halves + 8
    __shared__ unsigned short sB[128 * 40];

    const int tid  = threadIdx.x;
    const int lane = tid & 31;
    const int wave = tid >> 5;
    const int lm = lane & 15, lh = lane >> 4;
    const int wm = wave & 1,  wn = wave >> 1;
    const int mbase = blockIdx.y * 128;
    const int nbase = blockIdx.x * 128;
    const int sr  = tid >> 1;                 // staging row 0..127
    const int shf = (tid & 1) * 16;           // staging half (16 elems)

    const v8f vzero = {0.f, 0.f, 0.f, 0.f, 0.f, 0.f, 0.f, 0.f};
    v8f acc[4][2];
#pragma unroll
    for (int i = 0; i < 4; ++i)
#pragma unroll
        for (int j = 0; j < 2; ++j) acc[i][j] = vzero;

    for (int k0 = 0; k0 < K; k0 += 32) {
        // ---- stage A tile (convert to bf16 if needed) ----
        if (A_BF16) {
            const unsigned short* src =
                (const unsigned short*)Ap + (size_t)(mbase + sr) * K + k0 + shf;
            *(uint4*)&sA[sr * 40 + shf]     = *(const uint4*)src;
            *(uint4*)&sA[sr * 40 + shf + 8] = *(const uint4*)(src + 8);
        } else {
            const float* src = (const float*)Ap + (size_t)(mbase + sr) * K + k0 + shf;
            float4 f0 = *(const float4*)(src + 0);
            float4 f1 = *(const float4*)(src + 4);
            float4 f2 = *(const float4*)(src + 8);
            float4 f3 = *(const float4*)(src + 12);
            Frag16 t;
            t.h[0]=f2bf(f0.x); t.h[1]=f2bf(f0.y); t.h[2]=f2bf(f0.z); t.h[3]=f2bf(f0.w);
            t.h[4]=f2bf(f1.x); t.h[5]=f2bf(f1.y); t.h[6]=f2bf(f1.z); t.h[7]=f2bf(f1.w);
            t.h[8]=f2bf(f2.x); t.h[9]=f2bf(f2.y); t.h[10]=f2bf(f2.z); t.h[11]=f2bf(f2.w);
            t.h[12]=f2bf(f3.x); t.h[13]=f2bf(f3.y); t.h[14]=f2bf(f3.z); t.h[15]=f2bf(f3.w);
            *(uint4*)&sA[sr * 40 + shf]     = t.u[0];
            *(uint4*)&sA[sr * 40 + shf + 8] = t.u[1];
        }
        // ---- stage B tile (fp32 weights -> bf16) ----
        {
            const float* src = Bw + (size_t)(nbase + sr) * K + k0 + shf;
            float4 f0 = *(const float4*)(src + 0);
            float4 f1 = *(const float4*)(src + 4);
            float4 f2 = *(const float4*)(src + 8);
            float4 f3 = *(const float4*)(src + 12);
            Frag16 t;
            t.h[0]=f2bf(f0.x); t.h[1]=f2bf(f0.y); t.h[2]=f2bf(f0.z); t.h[3]=f2bf(f0.w);
            t.h[4]=f2bf(f1.x); t.h[5]=f2bf(f1.y); t.h[6]=f2bf(f1.z); t.h[7]=f2bf(f1.w);
            t.h[8]=f2bf(f2.x); t.h[9]=f2bf(f2.y); t.h[10]=f2bf(f2.z); t.h[11]=f2bf(f2.w);
            t.h[12]=f2bf(f3.x); t.h[13]=f2bf(f3.y); t.h[14]=f2bf(f3.z); t.h[15]=f2bf(f3.w);
            *(uint4*)&sB[sr * 40 + shf]     = t.u[0];
            *(uint4*)&sB[sr * 40 + shf + 8] = t.u[1];
        }
        // prefetch next tiles into cache hierarchy
        if (k0 + 32 < K) {
            __builtin_prefetch(Bw + (size_t)(nbase + sr) * K + k0 + 32 + shf, 0, 1);
            if (!A_BF16)
                __builtin_prefetch((const float*)Ap + (size_t)(mbase + sr) * K + k0 + 32 + shf, 0, 1);
        }
        __syncthreads();

        // ---- compute: 4 A-frags x 2 B-frags -> 8 WMMAs ----
        Frag16 af[4], bfr[2];
#pragma unroll
        for (int i = 0; i < 4; ++i) {
            const unsigned short* p = &sA[(wm * 64 + i * 16 + lm) * 40 + lh * 8];
            af[i].u[0] = *(const uint4*)p;
            af[i].u[1] = *(const uint4*)(p + 16);
        }
#pragma unroll
        for (int j = 0; j < 2; ++j) {
            const unsigned short* p = &sB[(wn * 32 + j * 16 + lm) * 40 + lh * 8];
            bfr[j].u[0] = *(const uint4*)p;
            bfr[j].u[1] = *(const uint4*)(p + 16);
        }
#pragma unroll
        for (int i = 0; i < 4; ++i)
#pragma unroll
            for (int j = 0; j < 2; ++j)
                acc[i][j] = wmma_bf16(af[i], bfr[j], acc[i][j]);
        __syncthreads();
    }

    // ---- store C ----
#pragma unroll
    for (int i = 0; i < 4; ++i)
#pragma unroll
        for (int j = 0; j < 2; ++j)
#pragma unroll
            for (int r = 0; r < 8; ++r) {
                const int m = mbase + wm * 64 + i * 16 + r + 8 * lh;
                const int n = nbase + wn * 32 + j * 16 + lm;
                if (OUT_BF16)
                    ((unsigned short*)Cp)[(size_t)m * N + n] = f2bf(acc[i][j][r]);
                else
                    ((float*)Cp)[(size_t)m * N + n] = acc[i][j][r];
            }
}

// ---------------------------------------------------------------------------
// In-place RoPE on bf16 QKV buffer; folds QSCALE into q heads.
// ---------------------------------------------------------------------------
__global__ __launch_bounds__(256) void rope_kernel(unsigned short* __restrict__ qkv,
                                                   const float* __restrict__ freqs) {
    int idx = blockIdx.x * 256 + threadIdx.x;          // SEQ * 24 heads * 128 pairs
    if (idx >= SEQ * 24 * 128) return;
    const int d    = idx & 127;
    const int rest = idx >> 7;
    const int hh   = rest % 24;
    const int t    = rest / 24;
    const int col  = (hh < NH) ? hh * HD : KOFF + (hh - NH) * HD;

    unsigned short* p = qkv + (size_t)t * NQKV + col + d;
    const float x1 = bf2f(p[0]);
    const float x2 = bf2f(p[128]);
    const float c  = freqs[(t * 128 + d) * 2 + 0];
    const float s  = freqs[(t * 128 + d) * 2 + 1];
    const float sc = (hh < NH) ? QSCALE : 1.f;
    p[0]   = f2bf((x1 * c - x2 * s) * sc);
    p[128] = f2bf((x1 * s + x2 * c) * sc);
}

// ---------------------------------------------------------------------------
// Flash attention with softcap + sliding-window causal mask.
// Block: 128 threads = 4 waves, each wave owns 16 queries of one head.
// Per 32-key tile: S^T = K*Q^T (lane-local softmax), then O += P*V with the
// probability A-fragment built lane-locally (A-layout == transposed C-layout).
// Fragment loads are double-buffered so each WMMA overlaps the next DS loads.
// ---------------------------------------------------------------------------
__global__ __launch_bounds__(128) void attn_fwd(const unsigned short* __restrict__ qkv,
                                                unsigned short* __restrict__ attn_out) {
    __shared__ unsigned short sK[32 * 264];    // 32 keys x 256 hd (+8 pad)
    __shared__ unsigned short sVT[256 * 40];   // 256 hd x 32 keys (+8 pad), transposed

    const int head = blockIdx.y;
    const int kv   = head >> 1;                // G = NH/NKV = 2
    const int q0b  = blockIdx.x * 64;
    const int tid  = threadIdx.x;
    const int lane = tid & 31, wave = tid >> 5;
    const int lm = lane & 15, lh = lane >> 4;
    const int q0w = q0b + wave * 16;

    // Q fragments (B-layout for S^T = K * Q^T): lane holds column q = q0w+lm
    Frag16 qf[8];
    {
        const unsigned short* qrow = qkv + (size_t)(q0w + lm) * NQKV + QOFF + head * HD;
#pragma unroll
        for (int c = 0; c < 8; ++c) {
            const int off = c * 32 + lh * 8;
            qf[c].u[0] = *(const uint4*)(qrow + off);
            qf[c].u[1] = *(const uint4*)(qrow + off + 16);
        }
    }

    const v8f vzero = {0.f, 0.f, 0.f, 0.f, 0.f, 0.f, 0.f, 0.f};
    v8f o[16];
#pragma unroll
    for (int c = 0; c < 16; ++c) o[c] = vzero;
    float mrun = NEGF;
    float lrun = 0.f;

    int lo = q0b - (WIN - 1);
    if (lo < 0) lo = 0;
    lo &= ~31;
    const int hi  = q0b + 64;
    const int wlo = q0w - (WIN - 1);

    // staging indices
    const int skey = tid & 31;                 // K staging: key row
    const int sdb  = (tid >> 5) * 64;          // K staging: d base (64 per wave)
    const int vkp  = tid & 15;                 // V staging: key pair (2*vkp, 2*vkp+1)
    const int vdb  = (tid >> 4) * 32;          // V staging: d base (32 per group)

    for (int t0 = lo; t0 < hi; t0 += 32) {
        // ---- cooperative staging ----
        {
            // K: row-major copy, b128 in / b128 out
            const unsigned short* krow =
                qkv + (size_t)(t0 + skey) * NQKV + KOFF + kv * HD + sdb;
#pragma unroll
            for (int i = 0; i < 8; ++i)
                *(uint4*)&sK[skey * 264 + sdb + i * 8] = *(const uint4*)(krow + i * 8);

            // V: transpose two keys at once, packing bf16 pairs -> ds_store_b32
            const unsigned short* va =
                qkv + (size_t)(t0 + 2 * vkp) * NQKV + VOFF + kv * HD + vdb;
            const unsigned short* vb = va + NQKV;       // next key row
            unsigned* sVT32 = (unsigned*)sVT;           // row stride 20 dwords
#pragma unroll
            for (int i = 0; i < 4; ++i) {
                uint4 a = *(const uint4*)(va + i * 8);
                uint4 b = *(const uint4*)(vb + i * 8);
                const int d = vdb + i * 8;
                sVT32[(d + 0) * 20 + vkp] = (a.x & 0xffffu) | (b.x << 16);
                sVT32[(d + 1) * 20 + vkp] = (a.x >> 16)     | (b.x & 0xffff0000u);
                sVT32[(d + 2) * 20 + vkp] = (a.y & 0xffffu) | (b.y << 16);
                sVT32[(d + 3) * 20 + vkp] = (a.y >> 16)     | (b.y & 0xffff0000u);
                sVT32[(d + 4) * 20 + vkp] = (a.z & 0xffffu) | (b.z << 16);
                sVT32[(d + 5) * 20 + vkp] = (a.z >> 16)     | (b.z & 0xffff0000u);
                sVT32[(d + 6) * 20 + vkp] = (a.w & 0xffffu) | (b.w << 16);
                sVT32[(d + 7) * 20 + vkp] = (a.w >> 16)     | (b.w & 0xffff0000u);
            }
        }
        __syncthreads();

        const bool active = (t0 <= q0w + 15) && (t0 + 31 >= wlo);
        if (active) {
            // ---- scores: S^T (keys x queries), two 16-key subtiles,
            //      double-buffered A-fragment loads ----
            v8f s0 = vzero, s1 = vzero;
            Frag16 ka[2], kb[2];
            {
                const unsigned short* pk0 = &sK[lm * 264 + lh * 8];
                const unsigned short* pk1 = &sK[(16 + lm) * 264 + lh * 8];
                ka[0].u[0] = *(const uint4*)pk0;
                ka[0].u[1] = *(const uint4*)(pk0 + 16);
                kb[0].u[0] = *(const uint4*)pk1;
                kb[0].u[1] = *(const uint4*)(pk1 + 16);
            }
#pragma unroll
            for (int c = 0; c < 8; ++c) {
                const int cur = c & 1, nxt = cur ^ 1;
                if (c < 7) {
                    const unsigned short* pk0 = &sK[lm * 264 + (c + 1) * 32 + lh * 8];
                    const unsigned short* pk1 = &sK[(16 + lm) * 264 + (c + 1) * 32 + lh * 8];
                    ka[nxt].u[0] = *(const uint4*)pk0;
                    ka[nxt].u[1] = *(const uint4*)(pk0 + 16);
                    kb[nxt].u[0] = *(const uint4*)pk1;
                    kb[nxt].u[1] = *(const uint4*)(pk1 + 16);
                }
                s0 = wmma_bf16(ka[cur], qf[c], s0);
                s1 = wmma_bf16(kb[cur], qf[c], s1);
            }

            // ---- softcap + mask + online softmax (query is lane-local) ----
            const int q = q0w + lm;
            float p0[8], p1[8];
            float rmax = NEGF;
#pragma unroll
            for (int r = 0; r < 8; ++r) {
                int t = t0 + lh * 8 + r;
                float sc = SOFTCAP * tanh_fast(s0[r] * (1.f / SOFTCAP));
                sc = (t <= q && t >= q - (WIN - 1)) ? sc : NEGF;
                p0[r] = sc; rmax = fmaxf(rmax, sc);
                t += 16;
                sc = SOFTCAP * tanh_fast(s1[r] * (1.f / SOFTCAP));
                sc = (t <= q && t >= q - (WIN - 1)) ? sc : NEGF;
                p1[r] = sc; rmax = fmaxf(rmax, sc);
            }
            rmax = fmaxf(rmax, __shfl_xor(rmax, 16, 32));
            const float mnew  = fmaxf(mrun, rmax);
            const float alpha = __expf(mrun - mnew);
            mrun = mnew;

            float lsum = 0.f;
            Frag16 pf;
#pragma unroll
            for (int r = 0; r < 8; ++r) {
                p0[r] = __expf(p0[r] - mnew); lsum += p0[r];
                p1[r] = __expf(p1[r] - mnew); lsum += p1[r];
                pf.h[r]     = f2bf(p0[r]);   // A-frag K = kb+r  (lane-local!)
                pf.h[8 + r] = f2bf(p1[r]);   // A-frag K = 16+kb+r
            }
            lsum += __shfl_xor(lsum, 16, 32);
            lrun = lrun * alpha + lsum;

            // rescale O rows (row q-local = r + 8*lh in C layout)
            float av[8];
#pragma unroll
            for (int r = 0; r < 8; ++r) av[r] = __shfl(alpha, r + 8 * lh, 32);
#pragma unroll
            for (int c = 0; c < 16; ++c)
#pragma unroll
                for (int r = 0; r < 8; ++r) o[c][r] = o[c][r] * av[r];

            // ---- O += P * V : 16 d-chunks, double-buffered B-frag loads ----
            Frag16 bv[2];
            {
                const unsigned short* vp = &sVT[lm * 40 + lh * 8];
                bv[0].u[0] = *(const uint4*)vp;
                bv[0].u[1] = *(const uint4*)(vp + 16);
            }
#pragma unroll
            for (int c = 0; c < 16; ++c) {
                const int cur = c & 1, nxt = cur ^ 1;
                if (c < 15) {
                    const unsigned short* vp = &sVT[((c + 1) * 16 + lm) * 40 + lh * 8];
                    bv[nxt].u[0] = *(const uint4*)vp;
                    bv[nxt].u[1] = *(const uint4*)(vp + 16);
                }
                o[c] = wmma_bf16(pf, bv[cur], o[c]);
            }
        }
        __syncthreads();
    }

    // ---- normalize and store bf16 ----
    float linv[8];
#pragma unroll
    for (int r = 0; r < 8; ++r) {
        const float l = __shfl(lrun, r + 8 * lh, 32);
        linv[r] = 1.f / l;
    }
#pragma unroll
    for (int c = 0; c < 16; ++c)
#pragma unroll
        for (int r = 0; r < 8; ++r) {
            const int qq = q0w + r + 8 * lh;
            attn_out[(size_t)qq * (NH * HD) + head * HD + c * 16 + lm] =
                f2bf(o[c][r] * linv[r]);
        }
}

// ---------------------------------------------------------------------------
// Launch
// ---------------------------------------------------------------------------
extern "C" void kernel_launch(void* const* d_in, const int* in_sizes, int n_in,
                              void* d_out, int out_size, void* d_ws, size_t ws_size,
                              hipStream_t stream) {
    (void)in_sizes; (void)n_in; (void)out_size; (void)ws_size;
    const float* hidden = (const float*)d_in[0];
    const float* freqs  = (const float*)d_in[1];
    // d_in[2..5] (kv_write_indices = arange, zero caches, causal mask) are
    // materialized analytically inside the kernels.
    const float* qkv_w  = (const float*)d_in[6];
    const float* o_w    = (const float*)d_in[7];
    float* out = (float*)d_out;

    char* ws = (char*)d_ws;
    unsigned short* qkv_bf  = (unsigned short*)ws;                                   // SEQ x 8192 bf16
    unsigned short* attn_bf = (unsigned short*)(ws + (size_t)SEQ * NQKV * 2);        // SEQ x 4096 bf16

    // 1) QKV projection: (3072 x 3584) x (8192 x 3584)^T -> bf16
    gemm_tn<false, true><<<dim3(NQKV / 128, SEQ / 128), 256, 0, stream>>>(
        hidden, qkv_w, qkv_bf, SEQ, NQKV, HID);

    // 2) RoPE in place (+ QSCALE folded into q)
    rope_kernel<<<(SEQ * 24 * 128 + 255) / 256, 256, 0, stream>>>(qkv_bf, freqs);

    // 3) Flash attention with softcap + sliding window
    attn_fwd<<<dim3(SEQ / 64, NH), 128, 0, stream>>>(qkv_bf, attn_bf);

    // 4) Output projection: (3072 x 4096) x (3584 x 4096)^T -> fp32
    gemm_tn<true, false><<<dim3(HID / 128, SEQ / 128), 256, 0, stream>>>(
        attn_bf, o_w, out, SEQ, HID, NH * HD);
}
